// Block_48395691492070
// MI455X (gfx1250) — compile-verified, matrix operand order checked by reading, not run on hardware
//
#include <hip/hip_runtime.h>
#include <hip/hip_bf16.h>

// ---------------------------------------------------------------------------
// Mamba block + FFN for gfx1250 (MI455X). All GEMMs via v_wmma_f32_16x16x32_bf16
// (wave32 WMMA). Double-buffered LDS with register-staged fp32->bf16
// conversion: global b128 loads for tile k+1 overlap WMMA compute on tile k.
// ---------------------------------------------------------------------------

#define D_MODEL  1024
#define D_STATE  16
#define D_CONV   4
#define D_INNER  2048
#define DT_RANK  64
#define BATCH    2
#define SEQLEN   1024
#define MTOK     (BATCH * SEQLEN)          // 2048 token rows

typedef __attribute__((ext_vector_type(16))) __bf16 v16bf;
typedef __attribute__((ext_vector_type(8)))  __bf16 v8bf;
typedef __attribute__((ext_vector_type(4)))  __bf16 v4bf;
typedef __attribute__((ext_vector_type(8)))  float  v8f;
typedef __attribute__((ext_vector_type(4)))  float  v4f;

// ---------------- GEMM: C[M,N] = A[M,K] * Bw[N,K]^T (+ epilogue) -----------
// Guarantees in this model: M % 128 == 0, K % 64 == 0, lda/ldb % 4 == 0.
// Only N may be ragged (clamped at load, guarded at store).
#define BM 128
#define BN 64
#define BK 64
#define LDSS 72   // LDS row stride (bf16 elems) = 144B; keeps b128 reads aligned

enum { EPI_NONE = 0, EPI_BIAS_SOFTPLUS = 1, EPI_RESID = 2,
       EPI_BIAS_RELU = 3, EPI_BIAS_RESID = 4 };

__device__ __forceinline__ v4bf cvt4(v4f v) {
  v4bf r;
  r.x = (__bf16)v.x; r.y = (__bf16)v.y; r.z = (__bf16)v.z; r.w = (__bf16)v.w;
  return r;
}

template <int EPI>
__global__ __launch_bounds__(256)
void gemm_bf16_wmma(const float* __restrict__ A, const float* __restrict__ Bw,
                    const float* __restrict__ bias, const float* __restrict__ resid,
                    float* __restrict__ C,
                    int M, int N, int K, int lda, int ldb, int ldc) {
  __shared__ __align__(16) __bf16 As[2][BM * LDSS];
  __shared__ __align__(16) __bf16 Bs[2][BN * LDSS];

  const int tid  = threadIdx.x;
  const int lane = tid & 31;
  const int wave = tid >> 5;
  const int wm   = wave & 3;            // 4 row-groups of 32
  const int wn   = wave >> 2;           // 2 col-groups of 32
  const int tileM = blockIdx.y * BM;
  const int tileN = blockIdx.x * BN;

  v8f acc[2][2] = {};

  const int fr   = lane & 15;
  const int koff = (lane < 16) ? 0 : 8; // ISA 16-bit A/B fragment layout

  // Staging geometry (float4 granularity): BK/4 = 16 float4 per row group.
  const int s_row  = tid >> 4;          // 16 rows per 256-thread pass
  const int s_col4 = tid & 15;          // float4 column in [0,16)

  // Distinct registers for in-flight global loads (A: 8, B: 4) so the
  // compiler can issue all 12 b128 loads before a single wait.
  v4f areg[8], breg[4];

  auto load_tiles = [&](int kk) {
#pragma unroll
    for (int i = 0; i < 8; ++i) {
      int row = i * 16 + s_row;
      areg[i] = *(const v4f*)(A + (size_t)(tileM + row) * lda + kk + s_col4 * 4);
    }
#pragma unroll
    for (int i = 0; i < 4; ++i) {
      int row = i * 16 + s_row;
      int gr  = min(tileN + row, N - 1);      // pad rows re-read row N-1
      breg[i] = *(const v4f*)(Bw + (size_t)gr * ldb + kk + s_col4 * 4);
    }
  };
  auto store_tiles = [&](int b) {
#pragma unroll
    for (int i = 0; i < 8; ++i) {
      int row = i * 16 + s_row;
      *(v4bf*)&As[b][row * LDSS + s_col4 * 4] = cvt4(areg[i]);
    }
#pragma unroll
    for (int i = 0; i < 4; ++i) {
      int row = i * 16 + s_row;
      *(v4bf*)&Bs[b][row * LDSS + s_col4 * 4] = cvt4(breg[i]);
    }
  };

  // Prologue: stage tile 0
  load_tiles(0);
  store_tiles(0);

  int buf = 0;
  for (int k0 = 0; k0 < K; k0 += BK) {
    __syncthreads();                          // LDS[buf] ready for reads
    const bool has_next = (k0 + BK) < K;
    if (has_next) load_tiles(k0 + BK);        // overlap HBM with WMMA below
    if (k0 + 2 * BK < K)                      // warm L2 one tile further out
      __builtin_prefetch(A + (size_t)(tileM + (tid >> 1)) * lda + k0 + 2 * BK +
                             (tid & 1) * 32, 0, 1);

    // ---- 2 WMMA K-steps of 32 within the BK=64 tile, from LDS[buf]
#pragma unroll
    for (int ks = 0; ks < BK; ks += 32) {
      v16bf bfrag[2];
#pragma unroll
      for (int tn = 0; tn < 2; ++tn) {
        const __bf16* bb = &Bs[buf][(wn * 32 + tn * 16 + fr) * LDSS + ks + koff];
        v8bf blo = *(const v8bf*)(bb);
        v8bf bhi = *(const v8bf*)(bb + 16);
        bfrag[tn] = __builtin_shufflevector(blo, bhi, 0, 1, 2, 3, 4, 5, 6, 7,
                                            8, 9, 10, 11, 12, 13, 14, 15);
      }
#pragma unroll
      for (int tm = 0; tm < 2; ++tm) {
        const __bf16* ab = &As[buf][(wm * 32 + tm * 16 + fr) * LDSS + ks + koff];
        v8bf alo = *(const v8bf*)(ab);
        v8bf ahi = *(const v8bf*)(ab + 16);
        v16bf afrag = __builtin_shufflevector(alo, ahi, 0, 1, 2, 3, 4, 5, 6, 7,
                                              8, 9, 10, 11, 12, 13, 14, 15);
#pragma unroll
        for (int tn = 0; tn < 2; ++tn) {
          acc[tm][tn] = __builtin_amdgcn_wmma_f32_16x16x32_bf16(
              false, afrag, false, bfrag[tn], (short)0, acc[tm][tn], false, false);
        }
      }
    }
    __syncthreads();                          // all waves done reading LDS[buf]
    if (has_next) {
      store_tiles(buf ^ 1);                   // wait_loadcnt lands here, post-WMMA
      buf ^= 1;
    }
  }

  // Epilogue + store per the 16x16 f32 C/D layout:
  // lane<16: (M=v, N=lane); lane>=16: (M=v+8, N=lane-16)
  const int nl = lane & 15;
  const int mo = (lane >> 4) * 8;
#pragma unroll
  for (int tm = 0; tm < 2; ++tm) {
#pragma unroll
    for (int tn = 0; tn < 2; ++tn) {
      int gn = tileN + wn * 32 + tn * 16 + nl;
      if (gn >= N) continue;
      float bv = (EPI == EPI_BIAS_SOFTPLUS || EPI == EPI_BIAS_RELU ||
                  EPI == EPI_BIAS_RESID) ? bias[gn] : 0.f;
#pragma unroll
      for (int v = 0; v < 8; ++v) {
        int gm = tileM + wm * 32 + tm * 16 + mo + v;
        float val = acc[tm][tn][v] + bv;
        if (EPI == EPI_BIAS_SOFTPLUS)
          val = (val > 20.f) ? val : log1pf(__expf(val));
        if (EPI == EPI_BIAS_RELU)
          val = fmaxf(val, 0.f);
        if (EPI == EPI_RESID || EPI == EPI_BIAS_RESID)
          val += resid[(size_t)gm * ldc + gn];
        C[(size_t)gm * ldc + gn] = val;
      }
    }
  }
}

// ---------------- LayerNorm over 1024 features, one block per row ----------
__global__ __launch_bounds__(256)
void ln_kernel(const float* __restrict__ x, const float* __restrict__ g,
               const float* __restrict__ b, float* __restrict__ out) {
  __shared__ float red[18];
  const int row = blockIdx.x;
  const float* xr = x + (size_t)row * D_MODEL;
  float s = 0.f, ss = 0.f;
  for (int i = threadIdx.x; i < D_MODEL; i += 256) {
    float v = xr[i];
    s += v;
    ss += v * v;
  }
#pragma unroll
  for (int o = 16; o > 0; o >>= 1) {
    s  += __shfl_xor(s, o, 32);
    ss += __shfl_xor(ss, o, 32);
  }
  const int wid = threadIdx.x >> 5;
  if ((threadIdx.x & 31) == 0) { red[wid] = s; red[8 + wid] = ss; }
  __syncthreads();
  if (threadIdx.x == 0) {
    float ts = 0.f, tss = 0.f;
    for (int w = 0; w < 8; ++w) { ts += red[w]; tss += red[8 + w]; }
    float mu  = ts / (float)D_MODEL;
    float var = tss / (float)D_MODEL - mu * mu;
    red[16] = mu;
    red[17] = rsqrtf(var + 1e-5f);
  }
  __syncthreads();
  const float mu = red[16], rinv = red[17];
  for (int i = threadIdx.x; i < D_MODEL; i += 256) {
    out[(size_t)row * D_MODEL + i] = (xr[i] - mu) * rinv * g[i] + b[i];
  }
}

// -------- causal depthwise conv1d (k=4) + bias + SiLU; xi = xz[:, :2048] ---
__global__ __launch_bounds__(256)
void conv_silu_kernel(const float* __restrict__ xz, const float* __restrict__ cw,
                      const float* __restrict__ cb, float* __restrict__ xc) {
  const int o = blockIdx.x * 256 + threadIdx.x;       // [0, B*L*D_INNER)
  const int d  = o & (D_INNER - 1);
  const int bt = o >> 11;                              // b*L + t
  const int t  = bt & (SEQLEN - 1);
  const int b  = bt >> 10;
  float acc = cb[d];
#pragma unroll
  for (int j = 0; j < D_CONV; ++j) {
    int tt = t - (D_CONV - 1) + j;
    if (tt >= 0) {
      float xv = xz[((size_t)(b * SEQLEN + tt)) * (2 * D_INNER) + d];
      acc = fmaf(cw[d * D_CONV + j], xv, acc);
    }
  }
  xc[(size_t)bt * D_INNER + d] = acc / (1.f + __expf(-acc));   // SiLU
}

// -------- selective scan: one lane per (b, d); 16-state kept in VGPRs ------
__global__ __launch_bounds__(256)
void scan_kernel(const float* __restrict__ dt, const float* __restrict__ x_dbl,
                 const float* __restrict__ xc, const float* __restrict__ A_log,
                 float* __restrict__ y) {
  const int g = blockIdx.x * 256 + threadIdx.x;  // [0, B*D_INNER)
  const int d = g & (D_INNER - 1);
  const int b = g >> 11;
  float Arow[D_STATE], h[D_STATE];
#pragma unroll
  for (int n = 0; n < D_STATE; ++n) {
    Arow[n] = -__expf(A_log[d * D_STATE + n]);
    h[n] = 0.f;
  }
  for (int t = 0; t < SEQLEN; ++t) {
    const int row = b * SEQLEN + t;
    const float dt_v = dt[(size_t)row * D_INNER + d];     // coalesced over d
    const float xc_v = xc[(size_t)row * D_INNER + d];
    const float* bc = x_dbl + (size_t)row * 96;           // [dt(64)|B(16)|C(16)]
    const float dbx = dt_v * xc_v;
    float acc = 0.f;
#pragma unroll
    for (int n = 0; n < D_STATE; ++n) {
      float dA = __expf(dt_v * Arow[n]);
      h[n] = fmaf(dA, h[n], dbx * bc[64 + n]);
      acc  = fmaf(h[n], bc[80 + n], acc);
    }
    y[(size_t)row * D_INNER + d] = acc;
  }
}

// -------- gate: yg = (y + D*xc) * silu(z); z = xz[:, 2048:] ----------------
__global__ __launch_bounds__(256)
void gate_kernel(const float* __restrict__ y, const float* __restrict__ xc,
                 const float* __restrict__ xz, const float* __restrict__ Dv,
                 float* __restrict__ yg) {
  const int o = blockIdx.x * 256 + threadIdx.x;
  const int d  = o & (D_INNER - 1);
  const int bt = o >> 11;
  float z = xz[(size_t)bt * (2 * D_INNER) + D_INNER + d];
  float sz = z / (1.f + __expf(-z));
  float v = fmaf(Dv[d], xc[o], y[o]);
  yg[o] = v * sz;
}

// ---------------------------------------------------------------------------
extern "C" void kernel_launch(void* const* d_in, const int* in_sizes, int n_in,
                              void* d_out, int out_size, void* d_ws, size_t ws_size,
                              hipStream_t stream) {
  (void)in_sizes; (void)n_in; (void)out_size; (void)ws_size;
  const float* x         = (const float*)d_in[0];
  const float* in_proj_w = (const float*)d_in[1];
  const float* conv_w    = (const float*)d_in[2];
  const float* conv_b    = (const float*)d_in[3];
  const float* x_proj_w  = (const float*)d_in[4];
  const float* dt_proj_w = (const float*)d_in[5];
  const float* dt_proj_b = (const float*)d_in[6];
  const float* A_log     = (const float*)d_in[7];
  const float* Dvec      = (const float*)d_in[8];
  const float* out_proj_w= (const float*)d_in[9];
  const float* ln1_g     = (const float*)d_in[10];
  const float* ln1_b     = (const float*)d_in[11];
  const float* ln2_g     = (const float*)d_in[12];
  const float* ln2_b     = (const float*)d_in[13];
  const float* ffn_w1    = (const float*)d_in[14];
  const float* ffn_b1    = (const float*)d_in[15];
  const float* ffn_w2    = (const float*)d_in[16];
  const float* ffn_b2    = (const float*)d_in[17];
  float* out = (float*)d_out;

  // Workspace carve-up (floats)
  float* ws = (float*)d_ws;
  float* xn    = ws;                       // 2048*1024
  float* xz    = xn    + MTOK * D_MODEL;   // 2048*4096
  float* xc    = xz    + MTOK * 2 * D_INNER;   // 2048*2048
  float* x_dbl = xc    + MTOK * D_INNER;   // 2048*96
  float* dt    = x_dbl + MTOK * 96;        // 2048*2048
  float* yb    = dt    + MTOK * D_INNER;   // 2048*2048
  float* yg    = yb    + MTOK * D_INNER;   // 2048*2048
  float* x1    = yg    + MTOK * D_INNER;   // 2048*1024
  float* h2    = x1    + MTOK * D_MODEL;   // 2048*1024
  float* f1    = h2    + MTOK * D_MODEL;   // 2048*4096

  const dim3 blk(256);
  const int elt_blocks = (MTOK * D_INNER) / 256;
  auto ggrid = [](int M, int N) { return dim3((N + BN - 1) / BN, (M + BM - 1) / BM); };

  // 1) LN1
  ln_kernel<<<MTOK, blk, 0, stream>>>(x, ln1_g, ln1_b, xn);
  // 2) in_proj: xz = xn @ in_proj_w^T            [2048,4096] K=1024
  gemm_bf16_wmma<EPI_NONE><<<ggrid(MTOK, 2 * D_INNER), blk, 0, stream>>>(
      xn, in_proj_w, nullptr, nullptr, xz,
      MTOK, 2 * D_INNER, D_MODEL, D_MODEL, D_MODEL, 2 * D_INNER);
  // 3) causal depthwise conv + SiLU
  conv_silu_kernel<<<elt_blocks, blk, 0, stream>>>(xz, conv_w, conv_b, xc);
  // 4) x_proj: x_dbl = xc @ x_proj_w^T           [2048,96] K=2048
  gemm_bf16_wmma<EPI_NONE><<<ggrid(MTOK, 96), blk, 0, stream>>>(
      xc, x_proj_w, nullptr, nullptr, x_dbl,
      MTOK, 96, D_INNER, D_INNER, D_INNER, 96);
  // 5) dt_proj + bias + softplus                 [2048,2048] K=64 (lda=96)
  gemm_bf16_wmma<EPI_BIAS_SOFTPLUS><<<ggrid(MTOK, D_INNER), blk, 0, stream>>>(
      x_dbl, dt_proj_w, dt_proj_b, nullptr, dt,
      MTOK, D_INNER, DT_RANK, 96, DT_RANK, D_INNER);
  // 6) selective scan
  scan_kernel<<<(BATCH * D_INNER) / 256, blk, 0, stream>>>(dt, x_dbl, xc, A_log, yb);
  // 7) gate
  gate_kernel<<<elt_blocks, blk, 0, stream>>>(yb, xc, xz, Dvec, yg);
  // 8) out_proj + residual(x): x1 = x + yg @ out_proj_w^T
  gemm_bf16_wmma<EPI_RESID><<<ggrid(MTOK, D_MODEL), blk, 0, stream>>>(
      yg, out_proj_w, nullptr, x, x1,
      MTOK, D_MODEL, D_INNER, D_INNER, D_INNER, D_MODEL);
  // 9) LN2
  ln_kernel<<<MTOK, blk, 0, stream>>>(x1, ln2_g, ln2_b, h2);
  // 10) FFN1 + bias + relu: f1 = relu(h2 @ ffn_w1^T + b1)  [2048,4096] K=1024
  gemm_bf16_wmma<EPI_BIAS_RELU><<<ggrid(MTOK, 4 * D_MODEL), blk, 0, stream>>>(
      h2, ffn_w1, ffn_b1, nullptr, f1,
      MTOK, 4 * D_MODEL, D_MODEL, D_MODEL, D_MODEL, 4 * D_MODEL);
  // 11) FFN2 + bias + residual(x1): out = x1 + f1 @ ffn_w2^T + b2
  gemm_bf16_wmma<EPI_BIAS_RESID><<<ggrid(MTOK, D_MODEL), blk, 0, stream>>>(
      f1, ffn_w2, ffn_b2, x1, out,
      MTOK, D_MODEL, 4 * D_MODEL, 4 * D_MODEL, 4 * D_MODEL, D_MODEL);
}
